// GQAInferenceAttention_58222576664986
// MI455X (gfx1250) — compile-verified, matrix operand order checked by reading, not run on hardware
//
#include <hip/hip_runtime.h>
#include <hip/hip_bf16.h>

#define HIDDEN   4096
#define N_HEADS  32
#define N_KV     8
#define HEAD_DIM 128
#define T_TOK    1024
#define BATCH    4
#define KV_DIM   (N_KV * HEAD_DIM)   // 1024

typedef __attribute__((ext_vector_type(16))) __bf16 v16bf;
typedef __attribute__((ext_vector_type(8)))  float  v8f;

// ---------------------------------------------------------------------------
// D = A(16x32 bf16) * B(32x16 bf16) + C(16x16 f32)
// ---------------------------------------------------------------------------
__device__ __forceinline__ v8f wmma_bf16(v16bf a, v16bf b, v8f c) {
  return __builtin_amdgcn_wmma_f32_16x16x32_bf16(
      /*neg_a=*/false, a, /*neg_b=*/false, b,
      /*c_mod=*/(short)0, c, /*reuse_a=*/false, /*reuse_b=*/false);
}

// Per-lane fragment loaders for the CDNA5 16-bit operand layout.
// khalf = (lane>=16) ? 8 : 0.  Fragment elements 0..7 hold K = khalf..khalf+7,
// elements 8..15 hold K = khalf+16..khalf+23 (ISA 7.12.2, 16-bit A 16x32).
// B (32x16, K-major per column) mirrors this with N = lane&15, so the same
// loader serves both operands whenever the data is contiguous along K.
__device__ __forceinline__ v16bf make_frag(const __bf16* p, int khalf) {
  union { uint4 u[2]; v16bf v; } r;
  r.u[0] = *(const uint4*)(p + khalf + 0);   // 8 bf16 = 16B
  r.u[1] = *(const uint4*)(p + khalf + 16);  // 8 bf16 = 16B
  return r.v;
}

// LDS-side loader (f32 probabilities -> bf16 A fragment)
__device__ __forceinline__ v16bf make_frag_f32(const float* p, int khalf) {
  float f[16];
  *(float4*)(f + 0)  = *(const float4*)(p + khalf + 0);
  *(float4*)(f + 4)  = *(const float4*)(p + khalf + 4);
  *(float4*)(f + 8)  = *(const float4*)(p + khalf + 16);
  *(float4*)(f + 12) = *(const float4*)(p + khalf + 20);
  v16bf r;
#pragma unroll
  for (int i = 0; i < 16; ++i) r[i] = (__bf16)f[i];
  return r;
}

// ---------------------------------------------------------------------------
// One-shot f32 -> bf16 conversion (8 elems/thread, 32B in / 16B out).
// Removes all v_cvt from the GEMM hot loops.
// ---------------------------------------------------------------------------
__global__ void __launch_bounds__(256)
cvt_bf16_kernel(const float* __restrict__ src, __bf16* __restrict__ dst) {
  size_t i = ((size_t)blockIdx.x * 256 + threadIdx.x) * 8;
  float4 a = *(const float4*)(src + i);
  float4 b = *(const float4*)(src + i + 4);
  union { uint4 u; __bf16 h[8]; } o;
  o.h[0] = (__bf16)a.x; o.h[1] = (__bf16)a.y;
  o.h[2] = (__bf16)a.z; o.h[3] = (__bf16)a.w;
  o.h[4] = (__bf16)b.x; o.h[5] = (__bf16)b.y;
  o.h[6] = (__bf16)b.z; o.h[7] = (__bf16)b.w;
  *(uint4*)(dst + i) = o.u;
}

// ---------------------------------------------------------------------------
// Pure-bf16 NT GEMM: C[m,n] = sum_k A[m,k] * W[n,k]  (A: MxK, W: NxK, row-major)
// Wave tile 64x64 (16 WMMAs / K-step, 8 fragment loads), 4 waves -> 128x128/WG.
// TRANSV: store C transposed as [m/Trows][n][m%Trows]  (used to build V^T).
// ---------------------------------------------------------------------------
template <typename OT, bool TRANSV>
__global__ void __launch_bounds__(128)
gemm_nt_kernel(const __bf16* __restrict__ A, const __bf16* __restrict__ W,
               OT* __restrict__ C, int N, int K, int Trows) {
  const int lane  = threadIdx.x & 31;
  const int wave  = threadIdx.x >> 5;
  const int Mbase = blockIdx.y * 128 + (wave & 1) * 64;
  const int Nbase = blockIdx.x * 128 + (wave >> 1) * 64;
  const int lrow  = lane & 15;
  const int khalf = (lane >> 4) << 3;   // 0 or 8

  const __bf16* arow[4];
  const __bf16* brow[4];
#pragma unroll
  for (int mt = 0; mt < 4; ++mt)
    arow[mt] = A + (size_t)(Mbase + mt * 16 + lrow) * K;
#pragma unroll
  for (int nt = 0; nt < 4; ++nt)
    brow[nt] = W + (size_t)(Nbase + nt * 16 + lrow) * K;

  v8f acc[4][4] = {};

  for (int k = 0; k < K; k += 32) {
    v16bf a[4], b[4];
#pragma unroll
    for (int mt = 0; mt < 4; ++mt) a[mt] = make_frag(arow[mt] + k, khalf);
#pragma unroll
    for (int nt = 0; nt < 4; ++nt) b[nt] = make_frag(brow[nt] + k, khalf);
#pragma unroll
    for (int mt = 0; mt < 4; ++mt)
#pragma unroll
      for (int nt = 0; nt < 4; ++nt)
        acc[mt][nt] = wmma_bf16(a[mt], b[nt], acc[mt][nt]);
  }

  // D layout: VGPR r -> row r (lanes 0-15) / r+8 (lanes 16-31); col = lane&15
  const int hi8 = khalf;
#pragma unroll
  for (int mt = 0; mt < 4; ++mt)
#pragma unroll
    for (int nt = 0; nt < 4; ++nt)
#pragma unroll
      for (int r = 0; r < 8; ++r) {
        int row = Mbase + mt * 16 + r + hi8;
        int col = Nbase + nt * 16 + (lane & 15);
        size_t idx;
        if (TRANSV)
          idx = ((size_t)((row / Trows) * N + col)) * Trows + (row % Trows);
        else
          idx = (size_t)row * N + col;
        C[idx] = (OT)acc[mt][nt][r];
      }
}

// ---------------------------------------------------------------------------
// Flash attention, GQA (4 query heads per KV head), causal tril(k=1) +
// padding mask.  One wave owns 16 query rows x D=128.  Q/K in (B,T,heads,128)
// bf16; V pre-transposed to [b][kv][d][t] so the PV B-operand is K-contiguous.
// ---------------------------------------------------------------------------
__global__ void __launch_bounds__(128)
gqa_flash_kernel(const __bf16* __restrict__ q, const __bf16* __restrict__ kx,
                 const __bf16* __restrict__ vT, const int* __restrict__ ids,
                 __bf16* __restrict__ attn) {
  __shared__ float pbuf[4][16 * 32];

  const int lane  = threadIdx.x & 31;
  const int wave  = threadIdx.x >> 5;
  const int qbase = blockIdx.x * 64 + wave * 16;
  const int h     = blockIdx.y;
  const int b     = blockIdx.z;
  const int kh    = h >> 2;                 // 32 heads -> 8 KV heads
  const int lrow  = lane & 15;
  const int khalf = (lane >> 4) << 3;       // 0 or 8
  const int hi8   = khalf;

  // Persistent Q fragments: 16 rows x 128 d = 4 K-chunks
  const __bf16* qrow =
      q + (size_t)(b * T_TOK + qbase + lrow) * HIDDEN + h * HEAD_DIM;
  v16bf qa[4];
#pragma unroll
  for (int dc = 0; dc < 4; ++dc) qa[dc] = make_frag(qrow + dc * 32, khalf);

  v8f   o[8] = {};
  float mrun[8], lrun[8];
#pragma unroll
  for (int r = 0; r < 8; ++r) { mrun[r] = -1.0e30f; lrun[r] = 0.0f; }

  const int*    idrow  = ids + b * T_TOK;
  const __bf16* kbase0 = kx + (size_t)(b * T_TOK) * KV_DIM + kh * HEAD_DIM;
  const __bf16* vbase0 = vT + (size_t)(b * N_KV + kh) * HEAD_DIM * T_TOK;
  float*        prow_w = &pbuf[wave][0];
  const float*  prow_r = &pbuf[wave][lrow * 32];

  const float inv_sqrt_d = 0.08838834764831845f;  // 1/sqrt(128)

  int nkt = (qbase + 16) / 32 + 1;                // causal limit (tril k=1)
  if (nkt > T_TOK / 32) nkt = T_TOK / 32;

  for (int kt = 0; kt < nkt; ++kt) {
    // ---- S = Q K^T for 32 keys (two 16-col tiles) ----
    v8f s0 = {}, s1 = {};
    {
      const __bf16* kr0 = kbase0 + (size_t)(kt * 32 + lrow) * KV_DIM;
      const __bf16* kr1 = kr0 + (size_t)16 * KV_DIM;
#pragma unroll
      for (int dc = 0; dc < 4; ++dc) {
        v16bf b0 = make_frag(kr0 + dc * 32, khalf);
        s0 = wmma_bf16(qa[dc], b0, s0);
        v16bf b1 = make_frag(kr1 + dc * 32, khalf);
        s1 = wmma_bf16(qa[dc], b1, s1);
      }
    }

    // ---- online softmax (row = VGPR index, cols across 16-lane halves) ----
    int  col0 = kt * 32 + (lane & 15);
    int  col1 = col0 + 16;
    bool pad0 = idrow[col0] != 0;
    bool pad1 = idrow[col1] != 0;
#pragma unroll
    for (int r = 0; r < 8; ++r) {
      int   qg = qbase + r + hi8;
      float v0 = s0[r] * inv_sqrt_d;
      float v1 = s1[r] * inv_sqrt_d;
      v0 = (pad0 && col0 <= qg + 1) ? v0 : -1.0e30f;
      v1 = (pad1 && col1 <= qg + 1) ? v1 : -1.0e30f;
      float mx = fmaxf(v0, v1);
#pragma unroll
      for (int off = 1; off < 16; off <<= 1)
        mx = fmaxf(mx, __shfl_xor(mx, off, 32));
      float mnew  = fmaxf(mrun[r], mx);
      float alpha = __expf(mrun[r] - mnew);
      float p0    = __expf(v0 - mnew);
      float p1    = __expf(v1 - mnew);
      float rs    = p0 + p1;
#pragma unroll
      for (int off = 1; off < 16; off <<= 1) rs += __shfl_xor(rs, off, 32);
      lrun[r] = lrun[r] * alpha + rs;
      mrun[r] = mnew;
      s0[r] = p0;
      s1[r] = p1;
#pragma unroll
      for (int d = 0; d < 8; ++d) o[d][r] *= alpha;
    }

    // ---- transpose P (D-layout -> A-layout) through per-wave LDS ----
#pragma unroll
    for (int r = 0; r < 8; ++r) {
      prow_w[(r + hi8) * 32 + (lane & 15)]      = s0[r];
      prow_w[(r + hi8) * 32 + 16 + (lane & 15)] = s1[r];
    }
    asm volatile("s_wait_dscnt 0" ::: "memory");   // wave-private LDS RAW
    v16bf pA = make_frag_f32(prow_r, khalf);
    asm volatile("s_wait_dscnt 0" ::: "memory");   // before next-iter stores

    // ---- O += P * V  (V^T rows are K-contiguous along keys) ----
    const __bf16* vstep = vbase0 + kt * 32;
#pragma unroll
    for (int nt = 0; nt < 8; ++nt) {
      int   drow = nt * 16 + (lane & 15);
      v16bf vb   = make_frag(vstep + (size_t)drow * T_TOK, khalf);
      o[nt] = wmma_bf16(pA, vb, o[nt]);
    }
  }

  // ---- finalize and store (B,T,H,D) == (B,T,4096), bf16 ----
#pragma unroll
  for (int r = 0; r < 8; ++r) {
    float invl = 1.0f / lrun[r];
    int   row  = qbase + r + hi8;
    __bf16* dst =
        attn + (size_t)(b * T_TOK + row) * HIDDEN + h * HEAD_DIM + (lane & 15);
#pragma unroll
    for (int nt = 0; nt < 8; ++nt) dst[nt * 16] = (__bf16)(o[nt][r] * invl);
  }
}

__global__ void write_pos_kernel(const int* __restrict__ spos,
                                 float* __restrict__ out_last) {
  if (threadIdx.x == 0) out_last[0] = (float)(spos[0] + T_TOK);
}

// ---------------------------------------------------------------------------
extern "C" void kernel_launch(void* const* d_in, const int* in_sizes, int n_in,
                              void* d_out, int out_size, void* d_ws,
                              size_t ws_size, hipStream_t stream) {
  (void)in_sizes; (void)n_in; (void)out_size; (void)ws_size;
  const int*   ids    = (const int*)d_in[0];
  const float* hidden = (const float*)d_in[1];
  const float* Wq     = (const float*)d_in[2];
  const float* Wk     = (const float*)d_in[3];
  const float* Wv     = (const float*)d_in[4];
  const float* Wo     = (const float*)d_in[5];
  const int*   spos   = (const int*)d_in[8];
  float*       out    = (float*)d_out;

  const int M = BATCH * T_TOK;  // 4096 tokens

  // Workspace layout (bytes).  attn reuses the hidden-bf16 slot (its last
  // reader is the V GEMM, which precedes the flash kernel in stream order);
  // a single 32MB slot is reused for each converted weight matrix.
  char* ws = (char*)d_ws;
  __bf16* hbf     = (__bf16*)(ws + 0);                       // 32 MB (then attn)
  __bf16* wbuf    = (__bf16*)(ws + (size_t)32 * 1024 * 1024);// 32 MB (weights)
  __bf16* q_ws    = (__bf16*)(ws + (size_t)64 * 1024 * 1024);// 32 MB
  __bf16* k_ws    = (__bf16*)(ws + (size_t)96 * 1024 * 1024);//  8 MB
  __bf16* vT_ws   = (__bf16*)(ws + (size_t)104 * 1024 * 1024);// 8 MB
  __bf16* attn_ws = hbf;

  const int nBig   = HIDDEN * HIDDEN;  // 16.7M  (hidden, Wq, Wo)
  const int nSmall = KV_DIM * HIDDEN;  //  4.2M  (Wk, Wv)
  dim3 blk(128);

  // one-shot bf16 conversions (bandwidth-trivial: ~250MB @ 23.3 TB/s)
  cvt_bf16_kernel<<<nBig / (256 * 8), 256, 0, stream>>>(hidden, hbf);
  cvt_bf16_kernel<<<nBig / (256 * 8), 256, 0, stream>>>(Wq, wbuf);
  // Q = hidden @ Wq^T  -> (B,T,32,128) bf16
  gemm_nt_kernel<__bf16, false>
      <<<dim3(HIDDEN / 128, M / 128), blk, 0, stream>>>(
          hbf, wbuf, q_ws, HIDDEN, HIDDEN, T_TOK);
  // K = hidden @ Wk^T  -> (B,T,8,128) bf16
  cvt_bf16_kernel<<<nSmall / (256 * 8), 256, 0, stream>>>(Wk, wbuf);
  gemm_nt_kernel<__bf16, false>
      <<<dim3(KV_DIM / 128, M / 128), blk, 0, stream>>>(
          hbf, wbuf, k_ws, KV_DIM, HIDDEN, T_TOK);
  // V = hidden @ Wv^T  -> transposed [b][kv][d][t] bf16 for K-contiguous PV
  cvt_bf16_kernel<<<nSmall / (256 * 8), 256, 0, stream>>>(Wv, wbuf);
  gemm_nt_kernel<__bf16, true>
      <<<dim3(KV_DIM / 128, M / 128), blk, 0, stream>>>(
          hbf, wbuf, vT_ws, KV_DIM, HIDDEN, T_TOK);
  // flash attention (GQA, causal k=1, padding mask); writes over hbf slot
  gqa_flash_kernel<<<dim3(T_TOK / 64, N_HEADS, BATCH), blk, 0, stream>>>(
      q_ws, k_ws, vT_ws, ids, attn_ws);
  // out = attn @ Wo^T  -> f32
  cvt_bf16_kernel<<<nBig / (256 * 8), 256, 0, stream>>>(Wo, wbuf);
  gemm_nt_kernel<float, false>
      <<<dim3(HIDDEN / 128, M / 128), blk, 0, stream>>>(
          attn_ws, wbuf, out, HIDDEN, HIDDEN, T_TOK);
  // second tuple element: start_pos + T
  write_pos_kernel<<<1, 1, 0, stream>>>(spos, out + (size_t)M * HIDDEN);
}
